// MHA_10505490006113
// MI455X (gfx1250) — matrix operand order repaired and checked by reading, not compiled, not run on hardware
//
#include <hip/hip_runtime.h>
#include <math.h>

#define cB 2
#define cS 2048
#define cDM 1024
#define cH 16
#define cHD 64
#define cC 64
#define cNC (cS / cC)            // 32 chunks
#define cM (cB * cS)             // 4096 rows
#define cNE (cB * cH * cS * cHD) // 4194304 per-head elems

typedef unsigned short u16;
typedef __attribute__((ext_vector_type(16))) __bf16 v16bf;
typedef __attribute__((ext_vector_type(8))) float v8f;
typedef __attribute__((ext_vector_type(4))) int v4i;

#define AS1 __attribute__((address_space(1)))
#define AS3 __attribute__((address_space(3)))

#if defined(__AMDGCN__) && __has_builtin(__builtin_amdgcn_global_load_async_to_lds_b128)
#define HAVE_ASYNC_LDS 1
#else
#define HAVE_ASYNC_LDS 0
#endif

// ---------- bf16 helpers (RNE) ----------
__device__ __forceinline__ u16 f2bf(float f) {
    unsigned int u = __float_as_uint(f);
    unsigned int r = u + 0x7FFFu + ((u >> 16) & 1u);
    return (u16)(r >> 16);
}
__device__ __forceinline__ float bf2f(u16 u) {
    return __uint_as_float(((unsigned int)u) << 16);
}

// ---------- WMMA fragment helpers (CDNA5 16x16x32 bf16 layouts) ----------
union FragU {
    v16bf v;
    uint4 q[2];
};

// A matrix 16x32 (MxK): lane l -> row (l&15); K chunks [(l>>4)*8 .. +7] and [+16 .. +23]
__device__ __forceinline__ FragU load_fragA(const u16* base, int ld) {
    const int lane = threadIdx.x & 31;
    const int row = lane & 15;
    const int kb = (lane >> 4) * 8;
    FragU f;
    const u16* p = base + row * ld + kb;
    f.q[0] = *(const uint4*)(p);
    f.q[1] = *(const uint4*)(p + 16);
    return f;
}
// B matrix 32x16 (KxN): lane l -> col (l&15); K chunks [(l>>4)*8 .. +7] and [+16 .. +23]
__device__ __forceinline__ FragU load_fragB(const u16* base, int ld) {
    const int lane = threadIdx.x & 31;
    const int col = lane & 15;
    const int kb = (lane >> 4) * 8;
    FragU f;
    const u16* p = base + col * ld + kb;
    f.q[0] = *(const uint4*)(p);
    f.q[1] = *(const uint4*)(p + 16);
    return f;
}

__device__ __forceinline__ v8f wmma_bf16(FragU a, FragU b, v8f c) {
    return __builtin_amdgcn_wmma_f32_16x16x32_bf16(
        /*neg_a=*/false, a.v, /*neg_b=*/false, b.v,
        /*c_mod=*/(short)0, c, /*reuse_a=*/false, /*reuse_b=*/false);
}

// 16B global -> LDS copy: async (CDNA5 GLOBAL_LOAD_ASYNC_TO_LDS_B128) if available
__device__ __forceinline__ void stage16(const u16* g, u16* l) {
#if HAVE_ASYNC_LDS
    __builtin_amdgcn_global_load_async_to_lds_b128((AS1 v4i*)g, (AS3 v4i*)l, 0, 0);
#else
    *(uint4*)l = *(const uint4*)g;
#endif
}

// ---------- shared 128x128 GEMM core ----------
// Computes acc[8] (one 16-row strip x 128 cols per wave) of  A[128,K] * W[128,K]^T
// Aw: this wave's A rows (m0*K), Wn: block's W rows (n0*K). Bl: 2*128*32 u16 LDS.
// B tile double-buffered in LDS (async-streamed); A frag and B frags double-buffered in regs.
__device__ __forceinline__ void gemm_128x128(const u16* __restrict__ Aw,
                                             const u16* __restrict__ Wn, int K,
                                             u16* Bl, v8f acc[8]) {
    const int tid = threadIdx.x;
    const int KT = K >> 5; // K-steps of 32
    const int brow = tid >> 1;       // 0..127: B tile row (n)
    const int bcol = (tid & 1) * 16; // 0 or 16 within 32-wide k slice
    // prologue: stage k-slice 0 into buffer 0  (2 async ops per wave)
    {
        const u16* g = Wn + (size_t)brow * K + bcol;
        u16* l = Bl + tid * 16;
        stage16(g, l);
        stage16(g + 8, l + 8);
    }
    FragU a_cur = load_fragA(Aw, K);
    for (int kt = 0; kt < KT; ++kt) {
        const int nk = (kt + 1) & (KT - 1); // wrap: last issue is redundant but harmless
        // prefetch next A frag first: in flight across stage + barrier + 8 WMMAs
        FragU a_nxt = load_fragA(Aw + nk * 32, K);
        { // stream next k-slice into the other buffer
            const u16* g = Wn + (size_t)brow * K + nk * 32 + bcol;
            u16* l = Bl + ((kt + 1) & 1) * (128 * 32) + tid * 16;
            stage16(g, l);
            stage16(g + 8, l + 8);
        }
#if HAVE_ASYNC_LDS
        asm volatile("s_wait_asynccnt 0x2" ::: "memory"); // current buffer's copies done
#endif
        __syncthreads();
        const u16* Bbuf = Bl + (kt & 1) * (128 * 32);
        // register-double-buffered B frags: load j+1 while MMA j executes
        FragU b_cur = load_fragB(Bbuf, 32);
#pragma unroll
        for (int j = 0; j < 8; ++j) {
            FragU b_nxt = b_cur;
            if (j < 7) b_nxt = load_fragB(Bbuf + (j + 1) * 16 * 32, 32);
            acc[j] = wmma_bf16(a_cur, b_cur, acc[j]);
            b_cur = b_nxt;
        }
        a_cur = a_nxt;
        __syncthreads(); // all waves done reading before this buffer is overwritten
    }
}

// ---------- kernel 0: fp32 -> bf16 cast ----------
__global__ void k_cast_bf16(const float* __restrict__ src, u16* __restrict__ dst, int n) {
    int i = blockIdx.x * blockDim.x + threadIdx.x;
    if (i < n) dst[i] = f2bf(src[i]);
}

// ---------- kernel 1: fused QKV GEMM + elu(.)+1 epilogue + head scatter ----------
__global__ void k_gemm_qkv(const u16* __restrict__ Xb, const u16* __restrict__ Wqkv,
                           const float* __restrict__ bqk, const float* __restrict__ bv,
                           u16* __restrict__ Qb, u16* __restrict__ Kb,
                           u16* __restrict__ Ktb, u16* __restrict__ Vtb) {
    __shared__ __align__(16) u16 Bl[2 * 128 * 32];
    const int wave = threadIdx.x >> 5;
    const int lane = threadIdx.x & 31;
    const int m0 = blockIdx.x * 128 + wave * 16;
    const int n0 = blockIdx.y * 128;
    v8f acc[8] = {};
    gemm_128x128(Xb + (size_t)m0 * cDM, Wqkv + (size_t)n0 * cDM, cDM, Bl, acc);

    const int hi = lane >> 4;
    const int nl = lane & 15;
#pragma unroll
    for (int j = 0; j < 8; ++j) {
        int n = n0 + j * 16 + nl;
        float bias = (n < 2 * cDM) ? bqk[n] : bv[n - 2 * cDM];
#pragma unroll
        for (int g = 0; g < 8; ++g) {
            int m = m0 + g + hi * 8;
            float val = acc[j][g] + bias;
            int bb = m >> 11; // / cS
            int s = m & (cS - 1);
            if (n < cDM) { // Q: elu+1
                int h = n >> 6, d = n & 63;
                float f = val > 0.f ? val + 1.f : __expf(val);
                Qb[((size_t)(bb * cH + h) * cS + s) * cHD + d] = f2bf(f);
            } else if (n < 2 * cDM) { // K: elu+1, plain + transposed
                int h = (n - cDM) >> 6, d = (n - cDM) & 63;
                float f = val > 0.f ? val + 1.f : __expf(val);
                u16 bf = f2bf(f);
                Kb[((size_t)(bb * cH + h) * cS + s) * cHD + d] = bf;
                Ktb[((size_t)(bb * cH + h) * cHD + d) * cS + s] = bf;
            } else { // V: transposed only
                int h = (n - 2 * cDM) >> 6, d = (n - 2 * cDM) & 63;
                Vtb[((size_t)(bb * cH + h) * cHD + d) * cS + s] = f2bf(val);
            }
        }
    }
}

// ---------- kernel 2: per-chunk stats  VKsum[t,d] = sum_s V[s,t]K[s,d]; ksum[d] ----------
__global__ void k_chunk_stats(const u16* __restrict__ Ktb, const u16* __restrict__ Vtb,
                              float* __restrict__ VKsum, float* __restrict__ ksum) {
    const int blk = blockIdx.x;
    const int c = blk & (cNC - 1);
    const int bh = blk >> 5;
    const int wave = threadIdx.x >> 5;
    const int lane = threadIdx.x & 31;
    const int t0 = wave * 16;
    const u16* Vbase = Vtb + ((size_t)bh * cHD + t0) * cS + c * cC; // A rows = t, K = s
    v8f acc[4] = {};
#pragma unroll
    for (int ks = 0; ks < cC; ks += 32) {
        FragU a = load_fragA(Vbase + ks, cS);
#pragma unroll
        for (int j = 0; j < 4; ++j) {
            FragU b = load_fragB(Ktb + ((size_t)bh * cHD + j * 16) * cS + c * cC + ks, cS);
            acc[j] = wmma_bf16(a, b, acc[j]);
        }
    }
    const int hi = lane >> 4, nl = lane & 15;
    float* out = VKsum + ((size_t)bh * cNC + c) * (cHD * cHD);
#pragma unroll
    for (int j = 0; j < 4; ++j)
#pragma unroll
        for (int g = 0; g < 8; ++g) {
            int t = t0 + g + hi * 8;
            int d = j * 16 + nl;
            out[t * cHD + d] = acc[j][g];
        }
    if (threadIdx.x < cHD) {
        int d = threadIdx.x;
        const u16* kr = Ktb + ((size_t)bh * cHD + d) * cS + c * cC;
        float sum = 0.f;
#pragma unroll
        for (int s2 = 0; s2 < cC; ++s2) sum += bf2f(kr[s2]);
        ksum[((size_t)bh * cNC + c) * cHD + d] = sum;
    }
}

// ---------- kernel 3: exclusive scan over chunks (KV state + k_cum state) ----------
__global__ void k_scan(const float* __restrict__ VKsum, const float* __restrict__ ksum,
                       u16* __restrict__ VKpre, float* __restrict__ kcumpre) {
    const int bh = blockIdx.x;
    const int e0 = threadIdx.x * 16; // 256 thr * 16 = 4096 elems
    float run[16];
#pragma unroll
    for (int i = 0; i < 16; ++i) run[i] = 0.f;
    for (int c = 0; c < cNC; ++c) {
        const float* src = VKsum + ((size_t)bh * cNC + c) * (cHD * cHD) + e0;
        u16* dst = VKpre + ((size_t)bh * cNC + c) * (cHD * cHD) + e0;
#pragma unroll
        for (int i = 0; i < 16; ++i) {
            dst[i] = f2bf(run[i]);
            run[i] += src[i];
        }
    }
    if (threadIdx.x < cHD) {
        int d = threadIdx.x;
        float r = 0.f;
        for (int c = 0; c < cNC; ++c) {
            kcumpre[((size_t)bh * cNC + c) * cHD + d] = r;
            r += ksum[((size_t)bh * cNC + c) * cHD + d];
        }
    }
}

// ---------- kernel 4: per-chunk attention ----------
// ctx[s,t] = (1/denom[s]) * ( sum_{s'<=s} P[s,s'] V[s',t] + sum_d Q[s,d] VKpre[t,d] )
// denom[s] = Q[s]·kcumpre + rowsum(tril(P));  P = Q K^T (intra-chunk)
__global__ void k_attn_chunk(const u16* __restrict__ Qb, const u16* __restrict__ Kb,
                             const u16* __restrict__ Vtb, const u16* __restrict__ VKpre,
                             const float* __restrict__ kcumpre, u16* __restrict__ ctx) {
    __shared__ __align__(16) u16 Pl[cC * cC];
    __shared__ float denom[cC];
    const int blk = blockIdx.x;
    const int c = blk & (cNC - 1);
    const int bh = blk >> 5;
    const int bb = bh >> 4;
    const int h = bh & 15;
    const int wave = threadIdx.x >> 5;
    const int lane = threadIdx.x & 31;
    const int hi = lane >> 4, nl = lane & 15;
    const u16* Qc = Qb + ((size_t)bh * cS + c * cC) * cHD; // rows m, ld=HD

    // denom0[m] = Q[m] . kcum_prev
    if (threadIdx.x < cC) {
        int m = threadIdx.x;
        const float* kc = kcumpre + ((size_t)bh * cNC + c) * cHD;
        const u16* qr = Qc + m * cHD;
        float sum = 0.f;
#pragma unroll
        for (int d = 0; d < cHD; ++d) sum += bf2f(qr[d]) * kc[d];
        denom[m] = sum;
    }
    __syncthreads();

    const int m0 = wave * 16;
    FragU qa0 = load_fragA(Qc + m0 * cHD, cHD);
    FragU qa1 = load_fragA(Qc + m0 * cHD + 32, cHD);

    // P = Q K^T (intra-chunk scores)
    v8f pacc[4] = {};
    const u16* Kc = Kb + ((size_t)bh * cS + c * cC) * cHD;
#pragma unroll
    for (int j = 0; j < 4; ++j) {
        FragU b0 = load_fragB(Kc + (j * 16) * cHD, cHD);
        pacc[j] = wmma_bf16(qa0, b0, pacc[j]);
        FragU b1 = load_fragB(Kc + (j * 16) * cHD + 32, cHD);
        pacc[j] = wmma_bf16(qa1, b1, pacc[j]);
    }
    // causal mask, row sums into denom, store P (bf16) to LDS for reuse as A operand
#pragma unroll
    for (int g = 0; g < 8; ++g) {
        int m = m0 + g + hi * 8;
        float rsum = 0.f;
#pragma unroll
        for (int j = 0; j < 4; ++j) {
            int n = j * 16 + nl;
            float v = (n <= m) ? pacc[j][g] : 0.f;
            rsum += v;
            Pl[m * cC + n] = f2bf(v);
        }
        rsum += __shfl_xor(rsum, 1, 16);
        rsum += __shfl_xor(rsum, 2, 16);
        rsum += __shfl_xor(rsum, 4, 16);
        rsum += __shfl_xor(rsum, 8, 16);
        if (nl == 0) denom[m] += rsum; // row uniquely owned by (wave,g,hi)
    }
    __syncthreads();
    if (threadIdx.x < cC) denom[threadIdx.x] = 1.f / denom[threadIdx.x];
    __syncthreads();

    // ctx = Q @ VKpre^T + P @ V
    v8f acc[4] = {};
    const u16* VKc = VKpre + ((size_t)bh * cNC + c) * (cHD * cHD); // rows t, ld=HD
    const u16* Vc = Vtb + (size_t)bh * cHD * cS + c * cC;          // rows t, ld=S
#pragma unroll
    for (int j = 0; j < 4; ++j) {
        FragU b0 = load_fragB(VKc + (j * 16) * cHD, cHD);
        acc[j] = wmma_bf16(qa0, b0, acc[j]);
        FragU b1 = load_fragB(VKc + (j * 16) * cHD + 32, cHD);
        acc[j] = wmma_bf16(qa1, b1, acc[j]);
    }
    FragU pa0 = load_fragA(Pl + m0 * cC, cC);
    FragU pa1 = load_fragA(Pl + m0 * cC + 32, cC);
#pragma unroll
    for (int j = 0; j < 4; ++j) {
        FragU b0 = load_fragB(Vc + (size_t)(j * 16) * cS, cS);
        acc[j] = wmma_bf16(pa0, b0, acc[j]);
        FragU b1 = load_fragB(Vc + (size_t)(j * 16) * cS + 32, cS);
        acc[j] = wmma_bf16(pa1, b1, acc[j]);
    }
    // normalize + scatter into [m, DM] layout for the output GEMM
#pragma unroll
    for (int j = 0; j < 4; ++j) {
        int t = j * 16 + nl;
#pragma unroll
        for (int g = 0; g < 8; ++g) {
            int m = m0 + g + hi * 8;
            float v = acc[j][g] * denom[m];
            ctx[((size_t)(bb * cS + c * cC + m)) * cDM + h * cHD + t] = f2bf(v);
        }
    }
}

// ---------- kernel 5: output GEMM, fp32 epilogue ----------
__global__ void k_gemm_out(const u16* __restrict__ Ab, const u16* __restrict__ Wb,
                           const float* __restrict__ bias, float* __restrict__ out) {
    __shared__ __align__(16) u16 Bl[2 * 128 * 32];
    const int wave = threadIdx.x >> 5;
    const int lane = threadIdx.x & 31;
    const int m0 = blockIdx.x * 128 + wave * 16;
    const int n0 = blockIdx.y * 128;
    v8f acc[8] = {};
    gemm_128x128(Ab + (size_t)m0 * cDM, Wb + (size_t)n0 * cDM, cDM, Bl, acc);

    const int hi = lane >> 4, nl = lane & 15;
#pragma unroll
    for (int j = 0; j < 8; ++j) {
        int n = n0 + j * 16 + nl;
        float bs = bias[n];
#pragma unroll
        for (int g = 0; g < 8; ++g) {
            int m = m0 + g + hi * 8;
            out[(size_t)m * cDM + n] = acc[j][g] + bs;
        }
    }
}

extern "C" void kernel_launch(void* const* d_in, const int* in_sizes, int n_in,
                              void* d_out, int out_size, void* d_ws, size_t ws_size,
                              hipStream_t stream) {
    const float* x = (const float*)d_in[0];
    const float* Wqk_w = (const float*)d_in[1];
    const float* Wqk_b = (const float*)d_in[2];
    const float* Wv_w = (const float*)d_in[3];
    const float* Wv_b = (const float*)d_in[4];
    const float* Wo_w = (const float*)d_in[5];
    const float* Wo_b = (const float*)d_in[6];
    float* out = (float*)d_out;

    // workspace carve-up (bf16 regions first, then fp32)
    u16* p = (u16*)d_ws;
    u16* Xb = p;    p += (size_t)cM * cDM;      // 4096x1024
    u16* Wqkv = p;  p += (size_t)3 * cDM * cDM; // 3072x1024
    u16* Wob = p;   p += (size_t)cDM * cDM;     // 1024x1024
    u16* Qb = p;    p += (size_t)cNE;           // [b,h,s,d]
    u16* Kb = p;    p += (size_t)cNE;           // [b,h,s,d]
    u16* Ktb = p;   p += (size_t)cNE;           // [b,h,d,s]
    u16* Vtb = p;   p += (size_t)cNE;           // [b,h,d,s]
    u16* VKpre = p; p += (size_t)cNE;           // [b,h,c,t,d] bf16
    u16* ctx = p;   p += (size_t)cM * cDM;      // [m, DM]
    float* fp = (float*)p;
    float* VKsum = fp;   fp += (size_t)cNE;     // [b,h,c,t,d] fp32
    float* ksum = fp;    fp += (size_t)cB * cH * cNC * cHD;
    float* kcumpre = fp; fp += (size_t)cB * cH * cNC * cHD;

    // 1) casts to bf16
    {
        int n1 = cM * cDM;
        k_cast_bf16<<<(n1 + 255) / 256, 256, 0, stream>>>(x, Xb, n1);
        int n2 = 2 * cDM * cDM;
        k_cast_bf16<<<(n2 + 255) / 256, 256, 0, stream>>>(Wqk_w, Wqkv, n2);
        int n3 = cDM * cDM;
        k_cast_bf16<<<(n3 + 255) / 256, 256, 0, stream>>>(Wv_w, Wqkv + (size_t)2 * cDM * cDM, n3);
        k_cast_bf16<<<(n3 + 255) / 256, 256, 0, stream>>>(Wo_w, Wob, n3);
    }
    // 2) fused QKV GEMM (M=4096, N=3072), async LDS-streamed weights, elu+1 + head scatter
    k_gemm_qkv<<<dim3(cM / 128, 3 * cDM / 128), 256, 0, stream>>>(
        Xb, Wqkv, Wqk_b, Wv_b, Qb, Kb, Ktb, Vtb);
    // 3) per-chunk KV outer-product sums (WMMA) + k sums
    k_chunk_stats<<<cB * cH * cNC, 128, 0, stream>>>(Ktb, Vtb, VKsum, ksum);
    // 4) exclusive prefix scan over chunks
    k_scan<<<cB * cH, 256, 0, stream>>>(VKsum, ksum, VKpre, kcumpre);
    // 5) per-chunk attention (intra tril(QK^T)V + inter Q·KVstate, normalized)
    k_attn_chunk<<<cB * cH * cNC, 128, 0, stream>>>(Qb, Kb, Vtb, VKpre, kcumpre, ctx);
    // 6) output projection GEMM (fp32 out)
    k_gemm_out<<<dim3(cM / 128, cDM / 128), 256, 0, stream>>>(ctx, Wob, Wo_b, out);
}
